// ScalingSelfAtt3D_65094524338380
// MI455X (gfx1250) — matrix operand order repaired and checked by reading, not compiled, not run on hardware
//
#include <hip/hip_runtime.h>
#include <hip/hip_bf16.h>
#include <math.h>

typedef __attribute__((ext_vector_type(4)))  float    vf4;
typedef __attribute__((ext_vector_type(2)))  float    vf2;
typedef __attribute__((ext_vector_type(16))) _Float16 v16h;
typedef __attribute__((ext_vector_type(8)))  float    v8f;
typedef __attribute__((ext_vector_type(8)))  unsigned v8u;

#define SPATIAL  1048576   // 64*128*128
#define SPLIT    16
#define CHUNK_V4 16384     // (SPATIAL/SPLIT)/4
#define NBC      256       // bs*C = 2*128

// ---------------------------------------------------------------------------
// Kernel 1: streaming partial reduction of x (max / sum / min per (b,c) chunk)
// 4096 blocks = 256 (b,c) pairs * 16 splits; 256 threads; 64 float4 per thread
// ---------------------------------------------------------------------------
__global__ __launch_bounds__(256)
void pool_partial_kernel(const float* __restrict__ x,
                         float* __restrict__ pmax,
                         float* __restrict__ psum,
                         float* __restrict__ pmin) {
  __shared__ float smax[256], ssum[256], smin[256];
  const int p   = blockIdx.x;      // 0..4095
  const int bc  = p >> 4;          // 0..255
  const int tid = threadIdx.x;
  const vf4* xv = (const vf4*)x;
  size_t base = (size_t)bc * (SPATIAL / 4) + (size_t)(p & 15) * CHUNK_V4;

  float mx = -3.4e38f, mn = 3.4e38f, sm = 0.f;
#pragma unroll 4
  for (int i = 0; i < 64; ++i) {
    vf4 v = __builtin_nontemporal_load(&xv[base + (size_t)i * 256 + tid]);
    mx = fmaxf(mx, fmaxf(fmaxf(v.x, v.y), fmaxf(v.z, v.w)));
    mn = fminf(mn, fminf(fminf(v.x, v.y), fminf(v.z, v.w)));
    sm += (v.x + v.y) + (v.z + v.w);
  }
  smax[tid] = mx; ssum[tid] = sm; smin[tid] = mn;
  __syncthreads();
  for (int off = 128; off > 0; off >>= 1) {
    if (tid < off) {
      smax[tid] = fmaxf(smax[tid], smax[tid + off]);
      smin[tid] = fminf(smin[tid], smin[tid + off]);
      ssum[tid] += ssum[tid + off];
    }
    __syncthreads();
  }
  if (tid == 0) { pmax[p] = smax[0]; psum[p] = ssum[0]; pmin[p] = smin[0]; }
}

// ---------------------------------------------------------------------------
// Kernel 2: finalize pooling -> xs (f16, padded 16x128); q/k/v projections via
// v_wmma_f32_16x16x32_f16 (8 waves, one 16-wide N tile each, shared A regs);
// then per-head softmax(16), attn*v, and both SE MLPs. One block, 256 threads.
// ---------------------------------------------------------------------------
__global__ __launch_bounds__(256)
void attn_se_kernel(const float* __restrict__ pmax, const float* __restrict__ psum,
                    const float* __restrict__ pmin,
                    const float* __restrict__ Wsq, const float* __restrict__ bsq,
                    const float* __restrict__ Wq,  const float* __restrict__ bq,
                    const float* __restrict__ Wk,  const float* __restrict__ bk,
                    const float* __restrict__ Wv,  const float* __restrict__ bv,
                    const float* __restrict__ s1w1, const float* __restrict__ s1b1,
                    const float* __restrict__ s1w2, const float* __restrict__ s1b2,
                    const float* __restrict__ s2w1, const float* __restrict__ s2b1,
                    const float* __restrict__ s2w2, const float* __restrict__ s2b2,
                    float* __restrict__ out) {
  __shared__ _Float16 xsh[16 * 128];   // A-matrix, batch padded 2 -> 16, rows 2..15 = 0
  __shared__ float qkv[3][2][128];     // [q/k/v][b][n]
  __shared__ float logits[256];
  __shared__ float outall[2 * 128];    // [b][head*16+i]  (first 64 = out1, next 64 = out2)
  __shared__ float hdn[64];            // [seBlock][b][mid]
  const int t = threadIdx.x;

  // ---- phase 1: combine the 16 partials per (b,c); xs -> f16 padded A matrix
  {
    float mx = -3.4e38f, mn = 3.4e38f, sm = 0.f;
    for (int s = 0; s < SPLIT; ++s) {
      int i = t * SPLIT + s;
      mx = fmaxf(mx, pmax[i]); mn = fminf(mn, pmin[i]); sm += psum[i];
    }
    float mean = sm * (1.0f / (float)SPATIAL);
    float v = mx * Wsq[0] + mean * Wsq[1] + mn * Wsq[2] + bsq[0];
    xsh[t] = (_Float16)v;                       // rows 0..1 (t = b*128 + c)
    for (int i = 256 + t; i < 16 * 128; i += 256) xsh[i] = (_Float16)0.f;  // pad rows
  }
  __syncthreads();

  // ---- phase 2: q,k,v = xs @ W.T + b via WMMA f32_16x16x32_f16
  {
    const int lane = t & 31;
    const int wave = t >> 5;         // n-tile 0..7
    const int half = lane >> 4;      // lane group
    const int row  = lane & 15;
    const int n    = wave * 16 + row;

    // Build A registers once (branch-free 32-bit LDS loads of f16 pairs).
    const unsigned* xs32 = (const unsigned*)xsh;
    v16h aReg[4];
#pragma unroll
    for (int kb = 0; kb < 4; ++kb) {
      v8u au;
#pragma unroll
      for (int v = 0; v < 8; ++v) {
        // A (16x32 f16) layout: K = 2*(v&3) + 8*half + 16*(v>>2)
        int kA = kb * 32 + 2 * (v & 3) + 8 * half + 16 * (v >> 2);
        au[v] = xs32[row * 64 + (kA >> 1)];
      }
      aReg[kb] = __builtin_bit_cast(v16h, au);
    }

    const float* Ws[3] = {Wq, Wk, Wv};
    const float* Bs[3] = {bq, bk, bv};
#pragma unroll
    for (int r = 0; r < 3; ++r) {
      const vf2* W2 = (const vf2*)Ws[r];
      v8f acc = {};
#pragma unroll
      for (int kb = 0; kb < 4; ++kb) {
        v16h bm;
#pragma unroll
        for (int v = 0; v < 8; ++v) {
          // B (32x16 f16) layout: K = 2*(v&3) + 8*half + 16*(v>>2)
          int kB = kb * 32 + 2 * (v & 3) + 8 * half + 16 * (v >> 2);
          vf2 w = W2[(n * 128 + kB) >> 1];
          bm[2 * v]     = (_Float16)w.x;
          bm[2 * v + 1] = (_Float16)w.y;
        }
        acc = __builtin_amdgcn_wmma_f32_16x16x32_f16(
            /*neg_a=*/false, aReg[kb], /*neg_b=*/false, bm,
            /*c_mod=*/(short)0, acc, /*reuse_a=*/false, /*reuse_b=*/false);
      }
      if (half == 0) {  // D: VGPR0 lanes0-15 = row M=0, VGPR1 = M=1 (real batch rows)
        qkv[r][0][n] = acc[0] + Bs[r][n];
        qkv[r][1][n] = acc[1] + Bs[r][n];
      }
    }
  }
  __syncthreads();

  // ---- phase 3: per-head softmax over 16 and attn * v
  {
    const int b    = t >> 7;
    const int head = (t >> 4) & 7;
    const int i    = t & 15;
    const int n    = head * 16 + i;
    float qv = qkv[0][b][n], kv = qkv[1][b][n], vv = qkv[2][b][n];
    float lg = qv * kv * 0.25f;     // scale = 16^-0.5
    logits[t] = lg;
    __syncthreads();
    const int base = t & ~15;
    float m = -3.4e38f;
    for (int j = 0; j < 16; ++j) m = fmaxf(m, logits[base + j]);
    float sume = 0.f;
    for (int j = 0; j < 16; ++j) sume += __expf(logits[base + j] - m);
    float attn = __expf(lg - m) / sume;
    if (head < 4) out[512 + b * 64 + head * 16 + i]       = attn;  // attn1
    else          out[640 + b * 64 + (head - 4) * 16 + i] = attn;  // attn2
    outall[b * 128 + n] = attn * vv;
  }
  __syncthreads();

  // ---- phase 4: SE hidden layers (relu(o @ w1.T + b1)), 64 values total
  if (t < 64) {
    const int blk = t >> 5, bb = (t >> 4) & 1, m = t & 15;
    const float* w1 = blk ? s2w1 : s1w1;
    const float* b1 = blk ? s2b1 : s1b1;
    const float* o  = &outall[bb * 128 + blk * 64];
    float acc = b1[m];
    for (int j = 0; j < 64; ++j) acc += o[j] * w1[m * 64 + j];
    hdn[blk * 32 + bb * 16 + m] = fmaxf(acc, 0.f);
  }
  __syncthreads();

  // ---- phase 5: SE outputs o1, o2 = sigmoid(hdn @ w2.T + b2)
  {
    const int bb = t >> 7, c = t & 127;
#pragma unroll
    for (int blk = 0; blk < 2; ++blk) {
      const float* w2 = blk ? s2w2 : s1w2;
      const float* b2 = blk ? s2b2 : s1b2;
      float acc = b2[c];
      for (int m = 0; m < 16; ++m) acc += hdn[blk * 32 + bb * 16 + m] * w2[c * 16 + m];
      out[blk * 256 + bb * 128 + c] = 1.f / (1.f + __expf(-acc));
    }
  }
}

// ---------------------------------------------------------------------------
extern "C" void kernel_launch(void* const* d_in, const int* in_sizes, int n_in,
                              void* d_out, int out_size, void* d_ws, size_t ws_size,
                              hipStream_t stream) {
  const float* x    = (const float*)d_in[0];
  const float* Wsq  = (const float*)d_in[1];
  const float* bsq  = (const float*)d_in[2];
  const float* Wq   = (const float*)d_in[3];
  const float* bq   = (const float*)d_in[4];
  const float* Wk   = (const float*)d_in[5];
  const float* bk   = (const float*)d_in[6];
  const float* Wv   = (const float*)d_in[7];
  const float* bv   = (const float*)d_in[8];
  const float* s1w1 = (const float*)d_in[9];
  const float* s1b1 = (const float*)d_in[10];
  const float* s1w2 = (const float*)d_in[11];
  const float* s1b2 = (const float*)d_in[12];
  const float* s2w1 = (const float*)d_in[13];
  const float* s2b1 = (const float*)d_in[14];
  const float* s2w2 = (const float*)d_in[15];
  const float* s2b2 = (const float*)d_in[16];
  float* out  = (float*)d_out;
  float* pmax = (float*)d_ws;              // 4096 floats
  float* psum = pmax + NBC * SPLIT;        // 4096 floats
  float* pmin = psum + NBC * SPLIT;        // 4096 floats

  pool_partial_kernel<<<NBC * SPLIT, 256, 0, stream>>>(x, pmax, psum, pmin);
  attn_se_kernel<<<1, 256, 0, stream>>>(pmax, psum, pmin, Wsq, bsq,
                                        Wq, bq, Wk, bk, Wv, bv,
                                        s1w1, s1b1, s1w2, s1b2,
                                        s2w1, s2b1, s2w2, s2b2, out);
}
